// Haea_592705487028
// MI455X (gfx1250) — compile-verified, matrix operand order checked by reading, not run on hardware
//
#include <hip/hip_runtime.h>
#include <math.h>

typedef __bf16 bf16;
typedef bf16  v16bf __attribute__((ext_vector_type(16)));
typedef bf16  v8bf  __attribute__((ext_vector_type(8)));
typedef float v8f   __attribute__((ext_vector_type(8)));
typedef float v4f   __attribute__((ext_vector_type(4)));

#define B_    4
#define TIME_ 32
#define NV_   24
#define D_    768
#define H_    12
#define DH_   64
#define NH_   4
#define BK_   64
#define L_    3
#define OUT_  768
#define S_    (TIME_*NV_)   /* 768  */
#define ST_   (2*S_)        /* 1536 */

// ---------------------------------------------------------------------------
// CDNA5 async global->LDS copy (GLOBAL_LOAD_ASYNC_TO_LDS_B128, ASYNCcnt).
// VDST = per-lane LDS byte offset, VADDR = 64-bit global address (GV mode).
// ---------------------------------------------------------------------------
__device__ inline void async_copy16(void* lds_dst, const void* gsrc) {
  unsigned loff = (unsigned)(size_t)lds_dst;            // LDS aperture low bits
  unsigned long long ga = (unsigned long long)(size_t)gsrc;
  asm volatile("global_load_async_to_lds_b128 %0, %1, off"
               :: "v"(loff), "v"(ga) : "memory");
}
__device__ inline void async_wait0() {
  asm volatile("s_wait_asynccnt 0x0" ::: "memory");
}

// ---------------------------------------------------------------------------
// WMMA fragment loaders (bf16 16x16x32, wave32).
// A-layout: lane(hf,m) holds K = hf*8..hf*8+7 and 16+hf*8..+7  ==> two
// contiguous 16-byte spans of a row-major [m][k] tile -> 2x ds_load_b128.
// B fragments use the same loader from [n][k]-layout tiles.
// ---------------------------------------------------------------------------
__device__ inline v16bf frag_ld_a(const bf16* p, int ld) {
  int lane = threadIdx.x & 31;
  int hf = lane >> 4, m = lane & 15;
  const bf16* b0 = p + m * ld + hf * 8;
  v8bf lo = *(const v8bf*)b0;
  v8bf hi = *(const v8bf*)(b0 + 16);
  return __builtin_shufflevector(lo, hi, 0, 1, 2, 3, 4, 5, 6, 7,
                                 8, 9, 10, 11, 12, 13, 14, 15);
}

__device__ inline v16bf frag_ld_a_f32(const float* p, int ld) {
  int lane = threadIdx.x & 31;
  int hf = lane >> 4, m = lane & 15;
  const float* b0 = p + m * ld + hf * 8;
  v4f a0 = *(const v4f*)b0;
  v4f a1 = *(const v4f*)(b0 + 4);
  v4f a2 = *(const v4f*)(b0 + 16);
  v4f a3 = *(const v4f*)(b0 + 20);
  v16bf r;
#pragma unroll
  for (int i = 0; i < 4; ++i) {
    r[i]      = (bf16)a0[i];
    r[4 + i]  = (bf16)a1[i];
    r[8 + i]  = (bf16)a2[i];
    r[12 + i] = (bf16)a3[i];
  }
  return r;
}

// ---------------------------------------------------------------------------
// fp32 -> bf16 transposing conversion [l][K][N] -> [l][N][K]
// ---------------------------------------------------------------------------
__global__ void k_cvt_t(const float* in, bf16* out, int K, int N, int nl) {
  long long idx = (long long)blockIdx.x * blockDim.x + threadIdx.x;
  long long kn = (long long)K * N;
  if (idx >= (long long)nl * kn) return;
  int l = (int)(idx / kn);
  int rem = (int)(idx % kn);
  int k = rem / N, n = rem % N;
  out[(size_t)l * kn + (size_t)n * K + k] = (bf16)in[idx];
}

// ---------------------------------------------------------------------------
// Embedding: mem = (src+ve)*sqrt(D); x = (tgt+ve+pe)*sqrt(D)
// ---------------------------------------------------------------------------
__global__ void k_embed(const float* src, const float* tgt, const float* ve,
                        float* mem, float* x) {
  int idx = blockIdx.x * blockDim.x + threadIdx.x;
  if (idx >= B_*S_*D_) return;
  int dd = idx % D_;
  int t  = (idx / D_) % S_;
  int var = t % NV_;
  int tm  = t / NV_;
  float v = ve[var * D_ + dd];
  int k = dd >> 1;
  float ang = (float)tm * expf(-(float)(2*k) * (logf(10000.f) / (float)D_));
  float pe = (dd & 1) ? cosf(ang) : sinf(ang);
  const float sc = 27.712812921102035f;  // sqrt(768)
  mem[idx] = (src[idx] + v) * sc;
  x[idx]   = (tgt[idx] + v + pe) * sc;
}

// ---------------------------------------------------------------------------
// LayerNorm, one wave32 per row; optional ReLU; batch row remap s_in -> s_out
// ---------------------------------------------------------------------------
__global__ void k_ln(const float* x, const float* g, const float* bta, float* out,
                     int rows, int cols, int relu, int s_in, int s_out) {
  int wave = threadIdx.x >> 5;
  int row = blockIdx.x * (blockDim.x >> 5) + wave;
  if (row >= rows) return;
  int lane = threadIdx.x & 31;
  const float* xr = x + (size_t)row * cols;
  float s = 0.f;
  for (int i = lane; i < cols; i += 32) s += xr[i];
  for (int m = 16; m > 0; m >>= 1) s += __shfl_xor(s, m, 32);
  float mean = s / (float)cols;
  float v = 0.f;
  for (int i = lane; i < cols; i += 32) { float d = xr[i] - mean; v += d * d; }
  for (int m = 16; m > 0; m >>= 1) v += __shfl_xor(v, m, 32);
  float inv = rsqrtf(v / (float)cols + 1e-5f);
  int orow = (row / s_in) * s_out + (row % s_in);
  float* op = out + (size_t)orow * cols;
  for (int i = lane; i < cols; i += 32) {
    float y = (xr[i] - mean) * inv * g[i] + bta[i];
    if (relu) y = fmaxf(y, 0.f);
    op[i] = y;
  }
}

// copy encoder memory into rows [S, ST) of the decoder h buffer
__global__ void k_copykv(const float* kvs, float* hbuf) {
  int idx = blockIdx.x * blockDim.x + threadIdx.x;
  if (idx >= B_*S_*D_) return;
  int d = idx % D_;
  int t = (idx / D_) % S_;
  int b = idx / (S_ * D_);
  hbuf[((size_t)b * ST_ + S_ + t) * D_ + d] = kvs[idx];
}

__global__ void k_add(float* x, const float* y, int n) {
  int i = blockIdx.x * blockDim.x + threadIdx.x;
  if (i < n) x[i] += y[i];
}

// GLU feed-forward activation: act = gelu(ug) * uv
__global__ void k_glu(const float* y1, float* act, int rows) {
  const int hn = 4 * D_;  // 3072
  int idx = blockIdx.x * blockDim.x + threadIdx.x;
  if (idx >= rows * hn) return;
  int m = idx / hn, j = idx % hn;
  float u = y1[(size_t)m * (8 * D_) + j];
  float gv = y1[(size_t)m * (8 * D_) + hn + j];
  float ge = 0.5f * u * (1.f + erff(u * 0.70710678118f));
  act[idx] = ge * gv;
}

// ---------------------------------------------------------------------------
// Tiled WMMA GEMM: C[M,N] = A[M,K](fp32 -> bf16 in LDS) x Bt[N,K](bf16, transposed)
// block = 256 threads (8 waves), tile 128x128, K-step 32.
// B tile staged via GLOBAL_LOAD_ASYNC_TO_LDS_B128 (ASYNCcnt).
// ---------------------------------------------------------------------------
__global__ void __launch_bounds__(256)
k_gemm(const float* A, const bf16* Bt, float* C, const float* bias,
       int M, int N, int K) {
  __shared__ alignas(16) bf16 As[128 * 32];  // [m][k]
  __shared__ alignas(16) bf16 Bs[128 * 32];  // [n][k]
  int row0 = blockIdx.y * 128, col0 = blockIdx.x * 128;
  int tid = threadIdx.x;
  int wave = tid >> 5, lane = tid & 31;
  int wm = wave & 3, wn = wave >> 2;
  v8f acc[2][4];
#pragma unroll
  for (int mi = 0; mi < 2; ++mi)
    for (int nt = 0; nt < 4; ++nt)
      for (int e = 0; e < 8; ++e) acc[mi][nt][e] = 0.f;

  for (int k0 = 0; k0 < K; k0 += 32) {
    // async-stage B (128 n-rows x 32 k), verbatim bf16 copies
    for (int ch = tid; ch < 512; ch += 256) {
      int n = ch >> 2, c = (ch & 3) * 8;
      async_copy16(&Bs[n * 32 + c], &Bt[(size_t)(col0 + n) * K + k0 + c]);
    }
    // stage A (128x32 fp32 -> bf16), 16-byte chunks
    for (int ch = tid; ch < 512; ch += 256) {
      int r = ch >> 2, c = (ch & 3) * 8;
      const float* srcp = &A[(size_t)(row0 + r) * K + k0 + c];
      v4f f0 = *(const v4f*)srcp;
      v4f f1 = *(const v4f*)(srcp + 4);
      v8bf o;
#pragma unroll
      for (int i = 0; i < 4; ++i) { o[i] = (bf16)f0[i]; o[4 + i] = (bf16)f1[i]; }
      *(v8bf*)&As[r * 32 + c] = o;
    }
    if (k0 + 32 < K) {  // prefetch next K tiles (global_prefetch_b8)
      __builtin_prefetch(&A[(size_t)(row0 + (tid & 127)) * K + k0 + 32], 0, 1);
      __builtin_prefetch(&Bt[(size_t)(col0 + (tid & 127)) * K + k0 + 32], 0, 1);
    }
    async_wait0();
    __syncthreads();
    v16bf a0 = frag_ld_a(&As[(wm * 32) * 32], 32);
    v16bf a1 = frag_ld_a(&As[(wm * 32 + 16) * 32], 32);
#pragma unroll
    for (int nt = 0; nt < 4; ++nt) {
      v16bf bf_ = frag_ld_a(&Bs[(wn * 64 + nt * 16) * 32], 32);
      acc[0][nt] = __builtin_amdgcn_wmma_f32_16x16x32_bf16(
          false, a0, false, bf_, (short)0, acc[0][nt], false, false);
      acc[1][nt] = __builtin_amdgcn_wmma_f32_16x16x32_bf16(
          false, a1, false, bf_, (short)0, acc[1][nt], false, false);
    }
    __syncthreads();
  }
  int hf = lane >> 4, nn = lane & 15;
#pragma unroll
  for (int mi = 0; mi < 2; ++mi)
    for (int nt = 0; nt < 4; ++nt)
      for (int r = 0; r < 8; ++r) {
        int rr = row0 + wm * 32 + mi * 16 + hf * 8 + r;
        int cc = col0 + wn * 64 + nt * 16 + nn;
        float v = acc[mi][nt][r];
        if (bias) v += bias[cc];
        C[(size_t)rr * N + cc] = v;
      }
}

// ---------------------------------------------------------------------------
// LSH bucket assignment: per (b,h,t) compute NH argmax over [rot, -rot]
// ---------------------------------------------------------------------------
__global__ void k_buckets(const float* qk, const float* rot, unsigned* okey,
                          int s_att, int nbh) {
  int gid = blockIdx.x * blockDim.x + threadIdx.x;
  int total = B_ * H_ * s_att;
  if (gid >= total) return;
  int t = gid % s_att;
  int bh = gid / s_att;
  int h = bh % H_, b = bh / H_;
  const float* q = qk + ((size_t)b * s_att + t) * D_ + h * DH_;
  int nb = 2 * nbh;
  for (int n = 0; n < NH_; ++n) {
    float best = -1e30f; int bi = 0;
    for (int r = 0; r < nbh; ++r) {
      float a = 0.f;
      for (int d = 0; d < DH_; ++d) a += q[d] * rot[(d * NH_ + n) * nbh + r];
      if (a > best)  { best = a;  bi = r; }
      if (-a > best) { best = -a; bi = r + nbh; }
    }
    int bucket = bi + n * nb;
    okey[(size_t)bh * (NH_ * s_att) + n * s_att + t] =
        (unsigned)(bucket * s_att + t);
  }
}

// ---------------------------------------------------------------------------
// Bitonic argsort in LDS per (b,h). key = order_key*8192 + ticker (fits u32)
// ---------------------------------------------------------------------------
__global__ void __launch_bounds__(1024)
k_sort(const unsigned* okey, int* sticker, int* undo, int n, int N2) {
  extern __shared__ unsigned sm[];
  int bh = blockIdx.x;
  const unsigned* keys = okey + (size_t)bh * n;
  for (int i = threadIdx.x; i < N2; i += blockDim.x)
    sm[i] = (i < n) ? (keys[i] * 8192u + (unsigned)i) : 0xFFFFFFFFu;
  __syncthreads();
  for (int k = 2; k <= N2; k <<= 1)
    for (int j = k >> 1; j > 0; j >>= 1) {
      for (int i = threadIdx.x; i < N2; i += blockDim.x) {
        int ixj = i ^ j;
        if (ixj > i) {
          unsigned a = sm[i], b = sm[ixj];
          bool up = ((i & k) == 0);
          if (up ? (a > b) : (a < b)) { sm[i] = b; sm[ixj] = a; }
        }
      }
      __syncthreads();
    }
  for (int i = threadIdx.x; i < n; i += blockDim.x) {
    int st = (int)(sm[i] & 8191u);
    sticker[(size_t)bh * n + i] = st;
    undo[(size_t)bh * n + st] = i;
  }
}

// ---------------------------------------------------------------------------
// Gather sorted q/k/v into chunk layout (bf16), normalize keys; one wave/row
// ---------------------------------------------------------------------------
__global__ void k_gather(const float* qk, const float* vv, const int* sticker,
                         bf16* bq, bf16* bk, bf16* bv, int* qpos, int s_att) {
  int row = blockIdx.x;  // [0, B*H*NH*s_att)
  int NHs = NH_ * s_att;
  int bh = row / NHs;
  int h = bh % H_, b = bh / H_;
  int t = sticker[row] % s_att;
  const float* qrow = qk + ((size_t)b * s_att + t) * D_ + h * DH_;
  const float* vrow = vv + ((size_t)b * s_att + t) * D_ + h * DH_;
  int lane = threadIdx.x;
  float q0 = qrow[lane], q1 = qrow[lane + 32];
  float ss = q0 * q0 + q1 * q1;
  for (int m = 16; m > 0; m >>= 1) ss += __shfl_xor(ss, m, 32);
  float inv = 1.0f / (sqrtf(ss) + 1e-9f);
  size_t base = (size_t)row * DH_;
  bq[base + lane]      = (bf16)q0;
  bq[base + lane + 32] = (bf16)q1;
  bk[base + lane]      = (bf16)(q0 * inv);
  bk[base + lane + 32] = (bf16)(q1 * inv);
  bv[base + lane]      = (bf16)vrow[lane];
  bv[base + lane + 32] = (bf16)vrow[lane + 32];
  if (lane == 0) qpos[row] = t;
}

// ---------------------------------------------------------------------------
// Chunked attention: 64 queries x 128 keys (chunk + previous chunk), WMMA bf16
// block = 128 threads (4 waves). q/k staged via async DMA; v staged transposed
// so P.V B-frags are contiguous; probs kept fp32 in LDS and cast in the loader.
// ---------------------------------------------------------------------------
__global__ void __launch_bounds__(128)
k_attn(const bf16* bq, const bf16* bk, const bf16* bv, const int* qpos,
       float* bo, float* lseg, int nc, int s_att, int use_mask) {
  __shared__ alignas(16) bf16 qs[64 * 64];    // [m][d]
  __shared__ alignas(16) bf16 ks[128 * 64];   // [j][d]  (== [n][k] for dots)
  __shared__ alignas(16) bf16 vst[64 * 128];  // [d][j]  (== [n][k] for P.V)
  __shared__ alignas(16) float ds[64 * 128];  // dots / probs
  __shared__ int qp[64];
  __shared__ int kp[128];
  int c = blockIdx.x % nc;
  int bh = blockIdx.x / nc;
  int cp = (c + nc - 1) % nc;
  size_t cbase = ((size_t)bh * nc + c) * 4096;
  size_t pbase = ((size_t)bh * nc + cp) * 4096;
  int tid = threadIdx.x;
  // async copies: q tile + both k tiles (verbatim global->LDS)
  for (int ch = tid; ch < 512; ch += 128) {
    int off = ch * 8;
    async_copy16(&qs[off], &bq[cbase + off]);
    async_copy16(&ks[off], &bk[cbase + off]);
    async_copy16(&ks[4096 + off], &bk[pbase + off]);
  }
  // v transpose into [d][j] (needs register shuffle -> sync path)
  for (int ch = tid; ch < 512; ch += 128) {
    int off = ch * 8;
    int j = off >> 6, d0 = off & 63;
    v8bf v0 = *(const v8bf*)&bv[cbase + off];
    v8bf v1 = *(const v8bf*)&bv[pbase + off];
#pragma unroll
    for (int e = 0; e < 8; ++e) {
      vst[(d0 + e) * 128 + j] = v0[e];
      vst[(d0 + e) * 128 + 64 + j] = v1[e];
    }
  }
  if (tid < 64) qp[tid] = qpos[(size_t)bh * nc * 64 + c * 64 + tid];
  if (tid < 128)
    kp[tid] = (tid < 64) ? qpos[(size_t)bh * nc * 64 + c * 64 + tid]
                         : qpos[(size_t)bh * nc * 64 + cp * 64 + (tid - 64)];
  async_wait0();
  __syncthreads();

  int wave = tid >> 5;
  int m0 = wave * 16;
  const float scale = 0.125f;  // DH^-0.5

  // dots = q . k^T
  for (int nt = 0; nt < 8; ++nt) {
    v8f acc;
    for (int e = 0; e < 8; ++e) acc[e] = 0.f;
#pragma unroll
    for (int kk = 0; kk < 64; kk += 32) {
      v16bf af = frag_ld_a(&qs[m0 * 64 + kk], 64);
      v16bf bfv = frag_ld_a(&ks[(nt * 16) * 64 + kk], 64);
      acc = __builtin_amdgcn_wmma_f32_16x16x32_bf16(
          false, af, false, bfv, (short)0, acc, false, false);
    }
    int lane = tid & 31, hf = lane >> 4, n = lane & 15;
    for (int r = 0; r < 8; ++r)
      ds[(m0 + hf * 8 + r) * 128 + nt * 16 + n] = acc[r] * scale;
  }
  __syncthreads();

  // masks
  for (int i = tid; i < 8192; i += 128) {
    int qi = i >> 7, kj = i & 127;
    float v = ds[i];
    int qpp = qp[qi], kpp = kp[kj];
    if (qpp == kpp) v = -1e5f;
    if (use_mask && qpp < S_ && kpp < S_ && (qpp / NV_) < (kpp / NV_))
      v += -1e9f;
    ds[i] = v;
  }
  __syncthreads();

  // row softmax + lse (2 threads per row, pair-shfl combine)
  {
    int row = tid >> 1, hh = tid & 1;
    float* dr = &ds[row * 128 + hh * 64];
    float mx = -1e30f;
    for (int j = 0; j < 64; ++j) mx = fmaxf(mx, dr[j]);
    mx = fmaxf(mx, __shfl_xor(mx, 1, 32));
    float sum = 0.f;
    for (int j = 0; j < 64; ++j) sum += expf(dr[j] - mx);
    sum += __shfl_xor(sum, 1, 32);
    float l = mx + logf(sum);
    for (int j = 0; j < 64; ++j) dr[j] = expf(dr[j] - l);
    if (hh == 0) lseg[(size_t)bh * nc * 64 + c * 64 + row] = l;
  }
  __syncthreads();

  // out = P . V
  for (int nt = 0; nt < 4; ++nt) {
    v8f acc;
    for (int e = 0; e < 8; ++e) acc[e] = 0.f;
#pragma unroll
    for (int kk = 0; kk < 128; kk += 32) {
      v16bf af = frag_ld_a_f32(&ds[m0 * 128 + kk], 128);
      v16bf bfv = frag_ld_a(&vst[(nt * 16) * 128 + kk], 128);
      acc = __builtin_amdgcn_wmma_f32_16x16x32_bf16(
          false, af, false, bfv, (short)0, acc, false, false);
    }
    int lane = tid & 31, hf = lane >> 4, n = lane & 15;
    for (int r = 0; r < 8; ++r)
      bo[cbase + (size_t)(m0 + hf * 8 + r) * 64 + nt * 16 + n] = acc[r];
  }
}

// ---------------------------------------------------------------------------
// Unsort + softmax-over-hash-rounds combine -> attn output [B*S, D]
// ---------------------------------------------------------------------------
__global__ void k_combine(const float* bo, const float* lseb, const int* undo,
                          float* attno, int s_att) {
  int gid = blockIdx.x * blockDim.x + threadIdx.x;
  if (gid >= B_ * H_ * S_ * DH_) return;
  int d = gid & 63;
  int t = (gid >> 6) % S_;
  int bh = gid / (DH_ * S_);
  int h = bh % H_, b = bh / H_;
  int NHs = NH_ * s_att;
  float ls[NH_]; int js[NH_];
  float mx = -1e30f;
  for (int n = 0; n < NH_; ++n) {
    int j = undo[(size_t)bh * NHs + n * s_att + t];
    js[n] = j;
    ls[n] = lseb[(size_t)bh * NHs + j];
    mx = fmaxf(mx, ls[n]);
  }
  float sum = 0.f, acc = 0.f;
  for (int n = 0; n < NH_; ++n) {
    float w = expf(ls[n] - mx);
    sum += w;
    acc += w * bo[((size_t)bh * NHs + js[n]) * DH_ + d];
  }
  attno[((size_t)(b * S_ + t)) * D_ + h * DH_ + d] = acc / sum;
}

// ---------------------------------------------------------------------------
// Host-side orchestration
// ---------------------------------------------------------------------------
struct Scratch {
  float *mem, *x, *hbuf, *qk, *vv, *attno, *gemmC, *act, *w2out;
  unsigned* okey;
  int *sticker, *undo, *qpos;
  bf16 *bq, *bk, *bv;
  float *bo, *lseb;
};

static void gemm(const float* A, const bf16* Bt, float* C, const float* bias,
                 int M, int N, int K, hipStream_t st) {
  k_gemm<<<dim3(N / 128, M / 128), dim3(256), 0, st>>>(A, Bt, C, bias, M, N, K);
}

static void run_layer(float* xio, const float* kv,
                      const float* ln1g, const float* ln1b,
                      const float* ln2g, const float* ln2b,
                      const bf16* bWqk, const bf16* bWv, const bf16* bWo,
                      const bf16* bW1, const bf16* bW2,
                      const float* b1, const float* b2,
                      const float* rot, int nbh, int use_mask,
                      Scratch& sc, hipStream_t st) {
  int s_att = kv ? ST_ : S_;
  int Mq = B_ * s_att;
  int NHs = NH_ * s_att;
  int nc = NHs / BK_;
  // h = LN1(x), concat kv for decoder
  k_ln<<<dim3((B_ * S_ + 7) / 8), dim3(256), 0, st>>>(
      xio, ln1g, ln1b, sc.hbuf, B_ * S_, D_, 0, S_, s_att);
  if (kv)
    k_copykv<<<dim3((B_ * S_ * D_ + 255) / 256), dim3(256), 0, st>>>(kv, sc.hbuf);
  gemm(sc.hbuf, bWqk, sc.qk, nullptr, Mq, D_, D_, st);
  gemm(sc.hbuf, bWv, sc.vv, nullptr, Mq, D_, D_, st);
  int tot = B_ * H_ * s_att;
  k_buckets<<<dim3((tot + 127) / 128), dim3(128), 0, st>>>(sc.qk, rot, sc.okey,
                                                           s_att, nbh);
  int N2 = 1; while (N2 < NHs) N2 <<= 1;
  k_sort<<<dim3(B_ * H_), dim3(1024), N2 * sizeof(unsigned), st>>>(
      sc.okey, sc.sticker, sc.undo, NHs, N2);
  k_gather<<<dim3(B_ * H_ * NHs), dim3(32), 0, st>>>(
      sc.qk, sc.vv, sc.sticker, sc.bq, sc.bk, sc.bv, sc.qpos, s_att);
  k_attn<<<dim3(B_ * H_ * nc), dim3(128), 0, st>>>(
      sc.bq, sc.bk, sc.bv, sc.qpos, sc.bo, sc.lseb, nc, s_att, use_mask);
  k_combine<<<dim3((B_ * H_ * S_ * DH_ + 255) / 256), dim3(256), 0, st>>>(
      sc.bo, sc.lseb, sc.undo, sc.attno, s_att);
  gemm(sc.attno, bWo, sc.gemmC, nullptr, B_ * S_, D_, D_, st);
  k_add<<<dim3((B_ * S_ * D_ + 255) / 256), dim3(256), 0, st>>>(
      xio, sc.gemmC, B_ * S_ * D_);
  // GLU FFN
  k_ln<<<dim3((B_ * S_ + 7) / 8), dim3(256), 0, st>>>(
      xio, ln2g, ln2b, sc.hbuf, B_ * S_, D_, 0, S_, S_);
  gemm(sc.hbuf, bW1, sc.gemmC, b1, B_ * S_, 8 * D_, D_, st);
  k_glu<<<dim3((B_ * S_ * 4 * D_ + 255) / 256), dim3(256), 0, st>>>(
      sc.gemmC, sc.act, B_ * S_);
  gemm(sc.act, bW2, sc.w2out, b2, B_ * S_, D_, 4 * D_, st);
  k_add<<<dim3((B_ * S_ * D_ + 255) / 256), dim3(256), 0, st>>>(
      xio, sc.w2out, B_ * S_ * D_);
}

static void cvt_t(const float* in, bf16* out, int K, int N, int nl,
                  hipStream_t st) {
  long long total = (long long)nl * K * N;
  k_cvt_t<<<dim3((unsigned)((total + 255) / 256)), dim3(256), 0, st>>>(
      in, out, K, N, nl);
}

extern "C" void kernel_launch(void* const* d_in, const int* in_sizes, int n_in,
                              void* d_out, int out_size, void* d_ws, size_t ws_size,
                              hipStream_t stream) {
  (void)in_sizes; (void)n_in; (void)out_size; (void)ws_size;
  const float* src     = (const float*)d_in[0];
  const float* tgt     = (const float*)d_in[1];
  const float* var_emb = (const float*)d_in[2];
  const float* e_Wqk = (const float*)d_in[3];
  const float* e_Wv  = (const float*)d_in[4];
  const float* e_Wo  = (const float*)d_in[5];
  const float* e_ln1g = (const float*)d_in[6];
  const float* e_ln1b = (const float*)d_in[7];
  const float* e_ln2g = (const float*)d_in[8];
  const float* e_ln2b = (const float*)d_in[9];
  const float* e_W1  = (const float*)d_in[10];
  const float* e_b1  = (const float*)d_in[11];
  const float* e_W2  = (const float*)d_in[12];
  const float* e_b2  = (const float*)d_in[13];
  const float* e_rot = (const float*)d_in[14];
  const float* d_Wqk = (const float*)d_in[15];
  const float* d_Wv  = (const float*)d_in[16];
  const float* d_Wo  = (const float*)d_in[17];
  const float* d_ln1g = (const float*)d_in[18];
  const float* d_ln1b = (const float*)d_in[19];
  const float* d_ln2g = (const float*)d_in[20];
  const float* d_ln2b = (const float*)d_in[21];
  const float* d_W1  = (const float*)d_in[22];
  const float* d_b1  = (const float*)d_in[23];
  const float* d_W2  = (const float*)d_in[24];
  const float* d_b2  = (const float*)d_in[25];
  const float* d_rot = (const float*)d_in[26];
  const float* o_W1  = (const float*)d_in[27];
  const float* o_b1  = (const float*)d_in[28];
  const float* o_lng = (const float*)d_in[29];
  const float* o_lnb = (const float*)d_in[30];
  const float* o_W2  = (const float*)d_in[31];
  const float* o_b2  = (const float*)d_in[32];

  // ---- workspace carving ----
  size_t cur = 0;
  char* ws = (char*)d_ws;
  auto carve = [&](size_t bytes) -> void* {
    void* p = ws + cur;
    cur += (bytes + 255) & ~(size_t)255;
    return p;
  };
  const size_t nQK = (size_t)L_ * D_ * D_;
  const size_t nW1 = (size_t)L_ * D_ * 8 * D_;
  const size_t nW2 = (size_t)L_ * 4 * D_ * D_;
  bf16* beWqk = (bf16*)carve(nQK * 2);
  bf16* beWv  = (bf16*)carve(nQK * 2);
  bf16* beWo  = (bf16*)carve(nQK * 2);
  bf16* beW1  = (bf16*)carve(nW1 * 2);
  bf16* beW2  = (bf16*)carve(nW2 * 2);
  bf16* bdWqk = (bf16*)carve(nQK * 2);
  bf16* bdWv  = (bf16*)carve(nQK * 2);
  bf16* bdWo  = (bf16*)carve(nQK * 2);
  bf16* bdW1  = (bf16*)carve(nW1 * 2);
  bf16* bdW2  = (bf16*)carve(nW2 * 2);
  bf16* boW1  = (bf16*)carve((size_t)D_ * OUT_ * 2);
  bf16* boW2  = (bf16*)carve((size_t)OUT_ * OUT_ * 2);

  Scratch sc;
  const size_t rows_max = (size_t)B_ * H_ * NH_ * ST_;  // 294912
  sc.mem    = (float*)carve((size_t)B_ * S_ * D_ * 4);
  sc.x      = (float*)carve((size_t)B_ * S_ * D_ * 4);
  sc.hbuf   = (float*)carve((size_t)B_ * ST_ * D_ * 4);
  sc.qk     = (float*)carve((size_t)B_ * ST_ * D_ * 4);
  sc.vv     = (float*)carve((size_t)B_ * ST_ * D_ * 4);
  sc.attno  = (float*)carve((size_t)B_ * S_ * D_ * 4);
  sc.gemmC  = (float*)carve((size_t)B_ * S_ * 8 * D_ * 4);
  sc.act    = (float*)carve((size_t)B_ * S_ * 4 * D_ * 4);
  sc.w2out  = (float*)carve((size_t)B_ * S_ * D_ * 4);
  sc.okey    = (unsigned*)carve(rows_max * 4);
  sc.sticker = (int*)carve(rows_max * 4);
  sc.undo    = (int*)carve(rows_max * 4);
  sc.qpos    = (int*)carve(rows_max * 4);
  sc.bq   = (bf16*)carve(rows_max * DH_ * 2);
  sc.bk   = (bf16*)carve(rows_max * DH_ * 2);
  sc.bv   = (bf16*)carve(rows_max * DH_ * 2);
  sc.bo   = (float*)carve(rows_max * DH_ * 4);
  sc.lseb = (float*)carve(rows_max * 4);

  // ---- convert weights to bf16, transposed to [N][K] ----
  cvt_t(e_Wqk, beWqk, D_, D_, L_, stream);
  cvt_t(e_Wv,  beWv,  D_, D_, L_, stream);
  cvt_t(e_Wo,  beWo,  D_, D_, L_, stream);
  cvt_t(e_W1,  beW1,  D_, 8 * D_, L_, stream);
  cvt_t(e_W2,  beW2,  4 * D_, D_, L_, stream);
  cvt_t(d_Wqk, bdWqk, D_, D_, L_, stream);
  cvt_t(d_Wv,  bdWv,  D_, D_, L_, stream);
  cvt_t(d_Wo,  bdWo,  D_, D_, L_, stream);
  cvt_t(d_W1,  bdW1,  D_, 8 * D_, L_, stream);
  cvt_t(d_W2,  bdW2,  4 * D_, D_, L_, stream);
  cvt_t(o_W1,  boW1,  D_, OUT_, 1, stream);
  cvt_t(o_W2,  boW2,  OUT_, OUT_, 1, stream);

  // ---- embeddings ----
  k_embed<<<dim3((B_ * S_ * D_ + 255) / 256), dim3(256), 0, stream>>>(
      src, tgt, var_emb, sc.mem, sc.x);

  // ---- encoder ----
  const int nbh_e = S_ / BK_ / 2;   // 6
  const int nbh_d = ST_ / BK_ / 2;  // 12
  for (int i = 0; i < L_; ++i) {
    run_layer(sc.mem, nullptr,
              e_ln1g + i * D_, e_ln1b + i * D_, e_ln2g + i * D_, e_ln2b + i * D_,
              beWqk + (size_t)i * D_ * D_, beWv + (size_t)i * D_ * D_,
              beWo + (size_t)i * D_ * D_, beW1 + (size_t)i * D_ * 8 * D_,
              beW2 + (size_t)i * 4 * D_ * D_,
              e_b1 + (size_t)i * 8 * D_, e_b2 + (size_t)i * D_,
              e_rot + (size_t)i * DH_ * NH_ * nbh_e, nbh_e, 0, sc, stream);
  }
  // ---- decoder ----
  for (int i = 0; i < L_; ++i) {
    run_layer(sc.x, sc.mem,
              d_ln1g + i * D_, d_ln1b + i * D_, d_ln2g + i * D_, d_ln2b + i * D_,
              bdWqk + (size_t)i * D_ * D_, bdWv + (size_t)i * D_ * D_,
              bdWo + (size_t)i * D_ * D_, bdW1 + (size_t)i * D_ * 8 * D_,
              bdW2 + (size_t)i * 4 * D_ * D_,
              d_b1 + (size_t)i * 8 * D_, d_b2 + (size_t)i * D_,
              d_rot + (size_t)i * DH_ * NH_ * nbh_d, nbh_d, 1, sc, stream);
  }
  // ---- output head ----
  gemm(sc.x, boW1, sc.gemmC, o_b1, B_ * S_, OUT_, D_, stream);
  k_ln<<<dim3((B_ * S_ + 7) / 8), dim3(256), 0, stream>>>(
      sc.gemmC, o_lng, o_lnb, sc.hbuf, B_ * S_, OUT_, 1, S_, S_);
  gemm(sc.hbuf, boW2, (float*)d_out, o_b2, B_ * S_, OUT_, OUT_, stream);
}
